// GatedDIP_39290360824440
// MI455X (gfx1250) — compile-verified, matrix operand order checked by reading, not run on hardware
//
#include <hip/hip_runtime.h>
#include <math.h>

typedef __attribute__((ext_vector_type(16))) _Float16 v16h;
typedef __attribute__((ext_vector_type(8)))  _Float16 v8h;
typedef __attribute__((ext_vector_type(8)))  float    v8f;

#define PLANES     24
#define PLANE_PIX  262144      // 512*512
#define OUT_PIX    6291456     // 24 * 262144
#define WS_GATE    0           // 16 floats
#define WS_CURVE   16          // 24 planes * (CP[8], dW[8], SL[8]) = 576 floats
#define WS_KW      592         // 24 planes * 98 floats (K1[49],K2[49]) = 2352
#define WS_MM      2944        // 3 stages * 24 planes * 2 uints = 144
#define PI_F       3.14159265358979323846f

// ---- monotonic float<->uint mapping (exact, order-preserving) ----
__device__ __forceinline__ unsigned f2mono(float f) {
  unsigned u = __float_as_uint(f);
  return (u & 0x80000000u) ? ~u : (u | 0x80000000u);
}
__device__ __forceinline__ float mono2f(unsigned u) {
  unsigned v = (u & 0x80000000u) ? (u & 0x7FFFFFFFu) : ~u;
  return __uint_as_float(v);
}

__device__ __forceinline__ float blockReduceMin(float v, float* rbuf) {
  int tid = threadIdx.x;
  rbuf[tid] = v; __syncthreads();
  for (int s = 128; s > 0; s >>= 1) {
    if (tid < s) rbuf[tid] = fminf(rbuf[tid], rbuf[tid + s]);
    __syncthreads();
  }
  float r = rbuf[0]; __syncthreads();
  return r;
}
__device__ __forceinline__ float blockReduceMax(float v, float* rbuf) {
  int tid = threadIdx.x;
  rbuf[tid] = v; __syncthreads();
  for (int s = 128; s > 0; s >>= 1) {
    if (tid < s) rbuf[tid] = fmaxf(rbuf[tid], rbuf[tid + s]);
    __syncthreads();
  }
  float r = rbuf[0]; __syncthreads();
  return r;
}

__device__ __forceinline__ float curve_apply(float xv, const float* cp) {
  float c = 0.0f;
#pragma unroll
  for (int l = 0; l < 8; ++l) {
    float d = fmaxf(xv - cp[l], 0.0f);
    d = fminf(d, cp[8 + l]);
    c += d * cp[16 + l];
  }
  return c;
}

// ---------------------------------------------------------------------------
// Kernel 1: three MLP heads via WMMA (layer1) + scalar layer2 + param derivation
// gridDim.x = 3 (head id), blockDim = 512 (16 wave32s)
// ---------------------------------------------------------------------------
__global__ void __launch_bounds__(512)
head_kernel(const float* __restrict__ f,
            const float* w1g, const float* b1g, const float* w2g, const float* b2g,
            const float* w1b, const float* b1b, const float* w2b, const float* b2b,
            const float* w1k, const float* b1k, const float* w2k, const float* b2k,
            float* __restrict__ ws, float* __restrict__ dout)
{
  __shared__ _Float16 sh_f[16 * 256]; // f16-staged A matrix, rows 8..15 zero
  __shared__ float    sh_h[8 * 256];  // layer-1 activations
  __shared__ float    sh_o[8 * 294];  // layer-2 outputs (max head dim)

  const int head = blockIdx.x;
  const int tid  = threadIdx.x;
  const float* w1 = (head == 0) ? w1g : (head == 1) ? w1b : w1k;
  const float* b1 = (head == 0) ? b1g : (head == 1) ? b1b : b1k;
  const float* w2 = (head == 0) ? w2g : (head == 1) ? w2b : w2k;
  const float* b2 = (head == 0) ? b2g : (head == 1) ? b2b : b2k;
  const int od    = (head == 0) ? 2  : (head == 1) ? 12  : 294;

  // ---- stage f into LDS as f16; pad rows 8..15 with zeros (branchless WMMA loads)
  {
    const int base = tid * 8;          // 4096 halfs / 512 threads
    const int row  = base >> 8;
    const int col  = base & 255;
    if (row < 8) {
      const float4 f0 = *(const float4*)(f + row * 256 + col);
      const float4 f1 = *(const float4*)(f + row * 256 + col + 4);
      sh_f[base + 0] = (_Float16)f0.x; sh_f[base + 1] = (_Float16)f0.y;
      sh_f[base + 2] = (_Float16)f0.z; sh_f[base + 3] = (_Float16)f0.w;
      sh_f[base + 4] = (_Float16)f1.x; sh_f[base + 5] = (_Float16)f1.y;
      sh_f[base + 6] = (_Float16)f1.z; sh_f[base + 7] = (_Float16)f1.w;
    } else {
#pragma unroll
      for (int j = 0; j < 8; ++j) sh_f[base + j] = (_Float16)0.0f;
    }
  }
  __syncthreads();

  // ---- layer 1: h[8x256] = leaky_relu(f @ w1^T + b1), via 16 waves of WMMA
  const int wv   = tid >> 5;     // wave id = N tile (0..15)
  const int lane = tid & 31;
  const int mn   = lane & 15;    // M index for A frag / N index for B frag
  const int hh   = lane >> 4;    // K-group half selector

  v8f acc = {};
  const float*    w1row = w1 + (wv * 16 + mn) * 256;  // B: lane holds w1[n, k...]
  const _Float16* arow  = sh_f + mn * 256;
#pragma unroll
  for (int kk = 0; kk < 8; ++kk) {
    const int kb = kk * 32 + hh * 8;
    // A fragment: two aligned 16B LDS loads (k = kb..kb+7 and kb+16..kb+23)
    const v8h alo = *(const v8h*)(arow + kb);
    const v8h ahi = *(const v8h*)(arow + kb + 16);
    v16h a;
#pragma unroll
    for (int e = 0; e < 8; ++e) { a[e] = alo[e]; a[8 + e] = ahi[e]; }
    // B fragment: four aligned 16B global loads + pack-convert to f16
    const float* bp = w1row + kb;
    const float4 b0 = *(const float4*)(bp);
    const float4 b1v = *(const float4*)(bp + 4);
    const float4 b2v = *(const float4*)(bp + 16);
    const float4 b3v = *(const float4*)(bp + 20);
    v16h bf;
    bf[0]  = (_Float16)b0.x;  bf[1]  = (_Float16)b0.y;
    bf[2]  = (_Float16)b0.z;  bf[3]  = (_Float16)b0.w;
    bf[4]  = (_Float16)b1v.x; bf[5]  = (_Float16)b1v.y;
    bf[6]  = (_Float16)b1v.z; bf[7]  = (_Float16)b1v.w;
    bf[8]  = (_Float16)b2v.x; bf[9]  = (_Float16)b2v.y;
    bf[10] = (_Float16)b2v.z; bf[11] = (_Float16)b2v.w;
    bf[12] = (_Float16)b3v.x; bf[13] = (_Float16)b3v.y;
    bf[14] = (_Float16)b3v.z; bf[15] = (_Float16)b3v.w;
    acc = __builtin_amdgcn_wmma_f32_16x16x32_f16(false, a, false, bf,
                                                 (short)0, acc, false, false);
  }
  // C/D layout: lanes 0..15 hold M=0..7 in vgprs 0..7, N = lane
  if (hh == 0) {
    const int n = wv * 16 + mn;
    const float bb = b1[n];
#pragma unroll
    for (int r = 0; r < 8; ++r) {
      float v = acc[r] + bb;
      sh_h[r * 256 + n] = (v > 0.0f) ? v : 0.2f * v;
    }
  }
  __syncthreads();

  // ---- layer 2: out2[8 x od] = h @ w2^T + b2 (scalar)
  for (int t = tid; t < 8 * od; t += 512) {
    const int m = t / od, o = t % od;
    const float* wr = w2 + o * 256;
    const float* hr = sh_h + m * 256;
    float s = b2[o];
    for (int k = 0; k < 256; ++k) s += hr[k] * wr[k];
    sh_o[t] = s;
  }
  __syncthreads();

  // ---- per-head derivations
  if (head == 0) {
    if (tid < 16) {
      float g = 1.0f / (1.0f + expf(-sh_o[tid]));
      g = fminf(fmaxf(g, 0.01f), 1.0f);
      ws[WS_GATE + tid] = g;
      dout[OUT_PIX + tid] = g;     // gate is second output, concatenated flat
    }
    // re-init min/max atomic slots each call
    if (tid < 144) {
      unsigned* mm = (unsigned*)(ws + WS_MM);
      mm[tid] = (tid & 1) ? 0u : 0xFFFFFFFFu;  // even=min slot, odd=max slot
    }
  } else if (head == 1) {
    if (tid < 24) {
      const int b = tid / 3, ch = tid % 3;
      float bp[4];
#pragma unroll
      for (int j = 0; j < 4; ++j) bp[j] = tanhf(sh_o[b * 12 + ch * 4 + j]);
      const float p1r = (bp[0] + 1.0f) * 0.5f, p2r = (bp[2] + 1.0f) * 0.5f;
      const float a1  = (bp[1] + 1.0f) * (PI_F * 0.25f);
      const float a2  = (bp[3] + 1.0f) * (PI_F * 0.25f);
      const float p1x = p1r * cosf(a1), p1y = p1r * sinf(a1);
      const float p2x = 1.0f - p2r * cosf(a2), p2y = 1.0f - p2r * sinf(a2);
      float Cx[9], Cy[9];
#pragma unroll
      for (int i = 0; i < 9; ++i) {
        const float q = (float)i * 0.125f, omq = 1.0f - q;
        const float B1v = 3.0f * omq * omq * q;
        const float B2v = 3.0f * omq * q * q;
        const float B3v = q * q * q;
        Cx[i] = B1v * p1x + B2v * p2x + B3v;
        Cy[i] = B1v * p1y + B2v * p2y + B3v;
      }
      float* dst = ws + WS_CURVE + tid * 24;   // plane = b*3+ch = tid
#pragma unroll
      for (int l = 0; l < 8; ++l) {
        const float dxp = fmaxf(Cx[l + 1] - Cx[l], 1e-6f);
        dst[l]      = Cx[l];
        dst[8 + l]  = dxp;
        dst[16 + l] = (Cy[l + 1] - Cy[l]) / dxp;
      }
    }
  } else {
    if (tid < 48) {
      const int b = tid / 6, kk = (tid % 6) / 3, ch = tid % 3;
      const int base = b * 294 + kk * 147 + ch * 49;
      float vals[49], s = 0.0f;
#pragma unroll
      for (int j = 0; j < 49; ++j) {
        float v = tanhf(sh_o[base + j]);
        vals[j] = v; s += fabsf(v);
      }
      const float inv = 1.0f / (s + 1e-6f);
      float* dst = ws + WS_KW + (b * 3 + ch) * 98 + kk * 49;
#pragma unroll
      for (int j = 0; j < 49; ++j) dst[j] = vals[j] * inv;
    }
  }
}

// ---------------------------------------------------------------------------
// Pass A: LDS-tiled depthwise conv -> t = x*conv1+conv2+x (stored in d_out),
//         curve value c; per-plane min/max of both via atomics.
// grid (1024, 24), block 256 (16x16 tile)
// ---------------------------------------------------------------------------
__global__ void __launch_bounds__(256)
passA(const float* __restrict__ x, const float* __restrict__ ws,
      float* __restrict__ dout)
{
  __shared__ float tile[22 * 22];
  __shared__ float kw[98];
  __shared__ float rbuf[256];

  const int plane = blockIdx.y;
  const int tx = blockIdx.x & 31, ty = blockIdx.x >> 5;
  const int lx = threadIdx.x & 15, ly = threadIdx.x >> 4;
  const float* xp = x + plane * PLANE_PIX;
  const int bx = tx * 16 - 3, by = ty * 16 - 3;

  for (int i = threadIdx.x; i < 484; i += 256) {
    const int r = i / 22, c = i % 22;
    const int gy = by + r, gx = bx + c;
    tile[i] = (gy >= 0 && gy < 512 && gx >= 0 && gx < 512) ? xp[gy * 512 + gx] : 0.0f;
  }
  if (threadIdx.x < 98) kw[threadIdx.x] = ws[WS_KW + plane * 98 + threadIdx.x];
  __syncthreads();

  const float xc = tile[(ly + 3) * 22 + lx + 3];
  float a1 = 0.0f, a2 = 0.0f;
#pragma unroll
  for (int i = 0; i < 7; ++i)
#pragma unroll
    for (int j = 0; j < 7; ++j) {
      const float v = tile[(ly + i) * 22 + lx + j];
      a1 += v * kw[i * 7 + j];
      a2 += v * kw[49 + i * 7 + j];
    }
  const float t = xc * a1 + a2 + xc;
  const float c = curve_apply(xc, ws + WS_CURVE + plane * 24);

  dout[plane * PLANE_PIX + (ty * 16 + ly) * 512 + (tx * 16 + lx)] = t;

  const float cmn = blockReduceMin(c, rbuf);
  const float cmx = blockReduceMax(c, rbuf);
  const float tmn = blockReduceMin(t, rbuf);
  const float tmx = blockReduceMax(t, rbuf);
  if (threadIdx.x == 0) {
    unsigned* mm = (unsigned*)(ws + WS_MM);
    atomicMin(&mm[0 * 48 + plane * 2 + 0], f2mono(cmn));
    atomicMax(&mm[0 * 48 + plane * 2 + 1], f2mono(cmx));
    atomicMin(&mm[1 * 48 + plane * 2 + 0], f2mono(tmn));
    atomicMax(&mm[1 * 48 + plane * 2 + 1], f2mono(tmx));
  }
}

// ---------------------------------------------------------------------------
// Pass B: s = norm(c)*g0 + norm(t)*g1; write s to d_out; min/max(s) -> stage2
// ---------------------------------------------------------------------------
__global__ void __launch_bounds__(256)
passB(const float* __restrict__ x, const float* __restrict__ ws,
      float* __restrict__ dout)
{
  __shared__ float rbuf[256];
  const int plane = blockIdx.y;
  const int idx = blockIdx.x * 256 + threadIdx.x;
  const int gidx = plane * PLANE_PIX + idx;

  const unsigned* mm = (const unsigned*)(ws + WS_MM);
  const float mnc = mono2f(mm[0 * 48 + plane * 2 + 0]);
  const float mxc = mono2f(mm[0 * 48 + plane * 2 + 1]);
  const float mnt = mono2f(mm[1 * 48 + plane * 2 + 0]);
  const float mxt = mono2f(mm[1 * 48 + plane * 2 + 1]);
  const int b = plane / 3;
  const float g0 = ws[WS_GATE + b * 2 + 0];
  const float g1 = ws[WS_GATE + b * 2 + 1];

  const float xv = x[gidx];
  const float c  = curve_apply(xv, ws + WS_CURVE + plane * 24);
  const float t  = dout[gidx];
  const float s  = (c - mnc) / (mxc - mnc + 1e-6f) * g0
                 + (t - mnt) / (mxt - mnt + 1e-6f) * g1;
  dout[gidx] = s;

  const float smn = blockReduceMin(s, rbuf);
  const float smx = blockReduceMax(s, rbuf);
  if (threadIdx.x == 0) {
    unsigned* mmw = (unsigned*)(ws + WS_MM);
    atomicMin(&mmw[2 * 48 + plane * 2 + 0], f2mono(smn));
    atomicMax(&mmw[2 * 48 + plane * 2 + 1], f2mono(smx));
  }
}

// ---------------------------------------------------------------------------
// Pass C: final in-place normalization of d_out
// ---------------------------------------------------------------------------
__global__ void __launch_bounds__(256)
passC(const float* __restrict__ ws, float* __restrict__ dout)
{
  const int plane = blockIdx.y;
  const int gidx = plane * PLANE_PIX + blockIdx.x * 256 + threadIdx.x;
  const unsigned* mm = (const unsigned*)(ws + WS_MM);
  const float mns = mono2f(mm[2 * 48 + plane * 2 + 0]);
  const float mxs = mono2f(mm[2 * 48 + plane * 2 + 1]);
  dout[gidx] = (dout[gidx] - mns) / (mxs - mns + 1e-6f);
}

extern "C" void kernel_launch(void* const* d_in, const int* in_sizes, int n_in,
                              void* d_out, int out_size, void* d_ws, size_t ws_size,
                              hipStream_t stream) {
  const float* x  = (const float*)d_in[0];
  const float* f  = (const float*)d_in[1];
  float* out = (float*)d_out;
  float* ws  = (float*)d_ws;

  head_kernel<<<3, 512, 0, stream>>>(f,
      (const float*)d_in[2],  (const float*)d_in[3],  (const float*)d_in[4],  (const float*)d_in[5],
      (const float*)d_in[6],  (const float*)d_in[7],  (const float*)d_in[8],  (const float*)d_in[9],
      (const float*)d_in[10], (const float*)d_in[11], (const float*)d_in[12], (const float*)d_in[13],
      ws, out);

  passA<<<dim3(1024, PLANES), 256, 0, stream>>>(x, ws, out);
  passB<<<dim3(1024, PLANES), 256, 0, stream>>>(x, ws, out);
  passC<<<dim3(1024, PLANES), 256, 0, stream>>>(ws, out);
}
